// SNN2_25125558682115
// MI455X (gfx1250) — matrix operand order repaired and checked by reading, not compiled, not run on hardware
//
#include <hip/hip_runtime.h>

typedef __attribute__((ext_vector_type(16))) _Float16 v16h;
typedef __attribute__((ext_vector_type(8)))  _Float16 v8h;
typedef __attribute__((ext_vector_type(2)))  _Float16 v2h;
typedef __attribute__((ext_vector_type(8)))  float    v8f;

#define T_STEPS 64
#define BETA    0.9f
#define THR     1.0f
// B=256, I=1024, H1=2048, H2=2048, O=256
#define S1_STRIDE 2056   // 2048 + 8 halves pad: row-to-row bank rotation of 4 -> conflict-free b128
#define DT_STRIDE 1032   // 1024 + 8 halves pad for the staged data tile

// LDS layout (279,040 B total, needs CDNA5's 320 KB WGP LDS):
//   [0      , 131072) : c1 tile, f32 [16][2048]           (static input current)
//   [131072 , 196864) : s1 spikes f16 [16][2056]  (also reused as data-f16 staging)
//   [196864 , 262656) : s2 spikes f16 [16][2056]
//   [262656 , 279040) : layer-3 partial sums, f32 [16 tiles][8][32 lanes]
__shared__ char smem_g[279040];

__device__ __forceinline__ v16h ldA(const _Float16* base, int mrow, int kb, int kc, int strideH) {
    const _Float16* p = base + mrow * strideH + kb + 16 * kc;  // K = 16c+h  (h<8)
    v8h lo = *(const v8h*)p;
    v8h hi = *(const v8h*)(p + 8);                             // K = 16c+8+h
    return __builtin_shufflevector(lo, hi, 0,1,2,3,4,5,6,7,8,9,10,11,12,13,14,15);
}

__global__ void wcvt(const float* __restrict__ W1, const float* __restrict__ W2,
                     const float* __restrict__ W3, _Float16* __restrict__ dst) {
    const int N1 = 2048 * 1024, N2 = 2048 * 2048, N3 = 256 * 2048;
    for (int i = blockIdx.x * blockDim.x + threadIdx.x; i < N1 + N2 + N3;
         i += gridDim.x * blockDim.x) {
        float v = (i < N1) ? W1[i] : (i < N1 + N2) ? W2[i - N1] : W3[i - N1 - N2];
        dst[i] = (_Float16)v;
    }
}

__global__ __launch_bounds__(1024, 1)
void snn_persistent(const float* __restrict__ data,
                    const float* __restrict__ b1,
                    const float* __restrict__ b2,
                    const float* __restrict__ b3,
                    const _Float16* __restrict__ W1h,
                    const _Float16* __restrict__ W2h,
                    const _Float16* __restrict__ W3h,
                    float* __restrict__ out) {
    float*    c1s = (float*)smem_g;
    _Float16* s1s = (_Float16*)(smem_g + 131072);
    _Float16* s2s = (_Float16*)(smem_g + 196864);
    _Float16* dth = (_Float16*)(smem_g + 131072);   // staging, dead after phase 1
    float*    p3s = (float*)(smem_g + 262656);      // layer-3 K-split partials

    const int tid  = threadIdx.x;
    const int lane = tid & 31;
    const int wv   = tid >> 5;          // wave 0..31
    const int r0   = blockIdx.x * 16;   // batch tile base row
    const int mrow = lane & 15;
    const int kc   = lane >> 4;

    // ---- Phase 0: stage this block's 16 data rows as f16 into LDS ----
    #pragma unroll
    for (int i = 0; i < 16; ++i)
        dth[i * DT_STRIDE + tid] = (_Float16)data[(r0 + i) * 1024 + tid];
    __syncthreads();

    // ---- Phase 1: c1 = data @ W1^T + b1  (WMMA f16, K=1024; 4 N-tiles/wave) ----
    {
        v8f acc[4]; int ngl[4];
        #pragma unroll
        for (int it = 0; it < 4; ++it) {
            ngl[it] = (wv * 4 + it) * 16 + mrow;
            float bv = b1[ngl[it]];
            #pragma unroll
            for (int j = 0; j < 8; ++j) acc[it][j] = bv;
        }
        for (int kb = 0; kb < 1024; kb += 32) {
            v16h a = ldA(dth, mrow, kb, kc, DT_STRIDE);
            v16h bf[4];
            #pragma unroll
            for (int it = 0; it < 4; ++it)          // 8 b128 loads in flight
                bf[it] = *(const v16h*)(W1h + (size_t)ngl[it] * 1024 + kb + 16 * kc);
            #pragma unroll
            for (int it = 0; it < 4; ++it)
                acc[it] = __builtin_amdgcn_wmma_f32_16x16x32_f16(
                              false, a, false, bf[it], (short)0, acc[it], false, false);
        }
        #pragma unroll
        for (int it = 0; it < 4; ++it)
            #pragma unroll
            for (int r = 0; r < 8; ++r)             // D: vgpr r, lanes 16-31 -> M = r+8
                c1s[(r + 8 * kc) * 2048 + ngl[it]] = acc[it][r];
    }
    __syncthreads();

    // ---- Persistent membrane state (registers for all 64 steps) ----
    float m1x[16], m1y[16];
    #pragma unroll
    for (int m = 0; m < 16; ++m) { m1x[m] = 0.f; m1y[m] = 0.f; }
    v8f m2[4]; v8f m3;
    #pragma unroll
    for (int it = 0; it < 4; ++it)
        #pragma unroll
        for (int j = 0; j < 8; ++j) m2[it][j] = 0.f;
    #pragma unroll
    for (int j = 0; j < 8; ++j) m3[j] = 0.f;

    float b2v[4]; int n2g[4];
    #pragma unroll
    for (int it = 0; it < 4; ++it) { n2g[it] = (wv * 4 + it) * 16 + mrow; b2v[it] = b2[n2g[it]]; }

    // layer-3 work split: wave = {tile 0..15} x {K-half}, all 32 waves busy
    const int t3   = wv & 15;           // output tile (cols t3*16 .. +15)
    const int kh3  = wv >> 4;           // 0: K=[0,1024)+state, 1: K=[1024,2048) partial
    const int n3g  = t3 * 16 + mrow;
    const float b3v = (kh3 == 0) ? b3[n3g] : 0.f;
    const int kbeg3 = kh3 * 1024;

    const int k0 = 2 * tid;   // layer-1 ownership: thread owns cols (2t,2t+1) of every row

    for (int t = 0; t < T_STEPS; ++t) {
        // ---- layer 1: fLIF on static current, emit s1 (f16 -> LDS) ----
        #pragma unroll
        for (int m = 0; m < 16; ++m) {
            float2 c = *(const float2*)(c1s + m * 2048 + k0);
            float vx = BETA * m1x[m] + c.x;
            float vy = BETA * m1y[m] + c.y;
            float sx = vx > THR ? 1.f : 0.f;
            float sy = vy > THR ? 1.f : 0.f;
            m1x[m] = vx - sx * THR;
            m1y[m] = vy - sy * THR;
            v2h s; s[0] = (_Float16)sx; s[1] = (_Float16)sy;
            *(v2h*)(s1s + m * S1_STRIDE + k0) = s;
        }
        __syncthreads();

        // ---- layer 2: m2 = beta*m2 + b2 + s1 @ W2^T  (WMMA chain into membrane) ----
        #pragma unroll
        for (int it = 0; it < 4; ++it)
            #pragma unroll
            for (int j = 0; j < 8; ++j) m2[it][j] = BETA * m2[it][j] + b2v[it];
        for (int kb = 0; kb < 2048; kb += 32) {
            v16h a = ldA(s1s, mrow, kb, kc, S1_STRIDE);
            v16h bf[4];
            #pragma unroll
            for (int it = 0; it < 4; ++it) {        // 8 b128 loads in flight
                const _Float16* bp = W2h + (size_t)n2g[it] * 2048 + kb + 16 * kc;
                __builtin_prefetch(bp + 128, 0, 1); // warm L0 four iterations ahead
                bf[it] = *(const v16h*)bp;
            }
            #pragma unroll
            for (int it = 0; it < 4; ++it)
                m2[it] = __builtin_amdgcn_wmma_f32_16x16x32_f16(
                             false, a, false, bf[it], (short)0, m2[it], false, false);
        }
        #pragma unroll
        for (int it = 0; it < 4; ++it)
            #pragma unroll
            for (int r = 0; r < 8; ++r) {
                float v = m2[it][r];
                float s = v > THR ? 1.f : 0.f;
                m2[it][r] = v - s * THR;
                s2s[(r + 8 * kc) * S1_STRIDE + n2g[it]] = (_Float16)s;
            }
        __syncthreads();

        // ---- layer 3: c3 = s2 @ W3^T + b3, K split across wave pairs ----
        {
            v8f acc;
            #pragma unroll
            for (int j = 0; j < 8; ++j)
                acc[j] = (kh3 == 0) ? (BETA * m3[j] + b3v) : 0.f;
            v16h bcur = *(const v16h*)(W3h + (size_t)n3g * 2048 + kbeg3 + 16 * kc);
            for (int kb = kbeg3; kb < kbeg3 + 1024; kb += 32) {
                v16h a = ldA(s2s, mrow, kb, kc, S1_STRIDE);
                v16h bnext = bcur;                  // double-buffered B
                if (kb + 32 < kbeg3 + 1024)
                    bnext = *(const v16h*)(W3h + (size_t)n3g * 2048 + kb + 32 + 16 * kc);
                acc = __builtin_amdgcn_wmma_f32_16x16x32_f16(
                          false, a, false, bcur, (short)0, acc, false, false);
                bcur = bnext;
            }
            if (kh3 == 1) {                         // publish partial sum
                #pragma unroll
                for (int r = 0; r < 8; ++r)
                    p3s[t3 * 256 + r * 32 + lane] = acc[r];
            }
            __syncthreads();
            if (kh3 == 0) {                         // combine, fLIF, write traces
                float* outS = out + t * 65536;              // spk_trace[t]
                float* outM = out + 4194304 + t * 65536;    // mem_trace[t]
                #pragma unroll
                for (int r = 0; r < 8; ++r) {
                    float v = acc[r] + p3s[t3 * 256 + r * 32 + lane];
                    float s = v > THR ? 1.f : 0.f;
                    m3[r] = v - s * THR;
                    int grow = r0 + r + 8 * kc;
                    outS[grow * 256 + n3g] = s;
                    outM[grow * 256 + n3g] = m3[r];
                }
            }
        }
        __syncthreads();
    }
}

extern "C" void kernel_launch(void* const* d_in, const int* in_sizes, int n_in,
                              void* d_out, int out_size, void* d_ws, size_t ws_size,
                              hipStream_t stream) {
    const float* data = (const float*)d_in[0];
    const float* W1   = (const float*)d_in[1];
    const float* b1   = (const float*)d_in[2];
    const float* W2   = (const float*)d_in[3];
    const float* b2   = (const float*)d_in[4];
    const float* W3   = (const float*)d_in[5];
    const float* b3   = (const float*)d_in[6];

    _Float16* wh = (_Float16*)d_ws;                     // 13,631,488 B of f16 weights
    wcvt<<<2048, 256, 0, stream>>>(W1, W2, W3, wh);
    const _Float16* W1h = wh;
    const _Float16* W2h = wh + 2048 * 1024;
    const _Float16* W3h = wh + 2048 * 1024 + 2048 * 2048;

    snn_persistent<<<16, 1024, 0, stream>>>(data, b1, b2, b3, W1h, W2h, W3h, (float*)d_out);
}
